// MAB_21603685499071
// MI455X (gfx1250) — matrix operand order repaired and checked by reading, not compiled, not run on hardware
//
#include <hip/hip_runtime.h>

typedef __attribute__((ext_vector_type(8)))  __bf16 v8bf;
typedef __attribute__((ext_vector_type(16))) __bf16 v16bf;
typedef __attribute__((ext_vector_type(8)))  float  v8f;
typedef __attribute__((ext_vector_type(4)))  int    v4i;

#define WMMA_BF16(a, b, c) \
  __builtin_amdgcn_wmma_f32_16x16x32_bf16(false, (a), false, (b), (short)0, (c), false, false)

static __device__ __forceinline__ v16bf cat8(v8bf lo, v8bf hi) {
  return __builtin_shufflevector(lo, hi, 0,1,2,3,4,5,6,7,8,9,10,11,12,13,14,15);
}

// ---- async global->LDS copy (CDNA5 GLOBAL_LOAD_ASYNC_TO_LDS_B128, ASYNCcnt)
#if __has_builtin(__builtin_amdgcn_global_load_async_to_lds_b128)
#define HAVE_ASYNC_LDS 1
static __device__ __forceinline__ void async_cp16(const __bf16* g, __bf16* l) {
  __builtin_amdgcn_global_load_async_to_lds_b128((v4i*)g, (v4i*)l, 0, 0);
}
#else
#define HAVE_ASYNC_LDS 0
#endif

static __device__ __forceinline__ void wait_async0() {
#if __has_builtin(__builtin_amdgcn_s_wait_asynccnt)
  __builtin_amdgcn_s_wait_asynccnt(0);
#else
  asm volatile("s_wait_asynccnt 0x0" ::: "memory");
#endif
}

// ---------------------------------------------------------------- converts
__global__ void k_f32_to_bf16(const float* __restrict__ s, __bf16* __restrict__ d, int n) {
  int i = blockIdx.x * blockDim.x + threadIdx.x;
  int stride = gridDim.x * blockDim.x;
  for (; i < n; i += stride) d[i] = (__bf16)s[i];
}

// ---------------------------------------------------------------- GEMM
// O[m,n] = epilogue( sum_k A[m,k] * W[n,k] + bias[n] )
// MODE 0: O = acc+bias             MODE 1: O = res + relu(acc+bias)
// Block: 256 thr (8 waves). Block tile 256x128, wave tile 64x64 (4x2 wave
// grid), K-step 32, double-buffered LDS with async global->LDS staging so the
// next tile's fetch overlaps the 16 WMMAs issued against the current tile.
template <int MODE, bool HASF, bool HASB>
__global__ void __launch_bounds__(256)
k_gemm(const __bf16* __restrict__ A, const __bf16* __restrict__ W,
       const float* __restrict__ bias, const float* __restrict__ res,
       float* __restrict__ outF, __bf16* __restrict__ outB,
       int M, int N, int K) {
  __shared__ alignas(16) __bf16 sA[2][256 * 40];
  __shared__ alignas(16) __bf16 sW[2][128 * 40];

  const int tid   = threadIdx.x;
  const int lane  = tid & 31;
  const int wave  = tid >> 5;
  const int wm    = wave >> 1;   // 0..3 (64-row slabs)
  const int wn    = wave & 1;    // 0..1 (64-col slabs)
  const int l16   = lane & 15;
  const int khalf = lane >> 4;

  const int rowBase = blockIdx.y * 256;
  const int colBase = blockIdx.x * 128;

  v8f acc[4][4];
  #pragma unroll
  for (int i = 0; i < 4; i++)
    #pragma unroll
    for (int j = 0; j < 4; j++)
      #pragma unroll
      for (int e = 0; e < 8; e++) acc[i][j][e] = 0.f;

  const int arow  = tid;        // 0..255, one full 32-elem A row per thread
  const int wrow  = tid >> 1;   // 0..127
  const int whalf = tid & 1;    // 16 elems each

  auto stage = [&](int k0s, int b) {
    const __bf16* ga = A + (size_t)(rowBase + arow) * K + k0s;
    const __bf16* gw = W + (size_t)(colBase + wrow) * K + k0s + whalf * 16;
    __bf16* bA = &sA[b][arow * 40];
    __bf16* bW = &sW[b][wrow * 40 + whalf * 16];
#if HAVE_ASYNC_LDS
    async_cp16(ga,      bA);
    async_cp16(ga + 8,  bA + 8);
    async_cp16(ga + 16, bA + 16);
    async_cp16(ga + 24, bA + 24);
    async_cp16(gw,      bW);
    async_cp16(gw + 8,  bW + 8);
#else
    v8bf a0 = *(const v8bf*)ga;        v8bf a1 = *(const v8bf*)(ga + 8);
    v8bf a2 = *(const v8bf*)(ga + 16); v8bf a3 = *(const v8bf*)(ga + 24);
    v8bf w0 = *(const v8bf*)gw;        v8bf w1 = *(const v8bf*)(gw + 8);
    *(v8bf*)(bA)      = a0;
    *(v8bf*)(bA + 8)  = a1;
    *(v8bf*)(bA + 16) = a2;
    *(v8bf*)(bA + 24) = a3;
    *(v8bf*)(bW)      = w0;
    *(v8bf*)(bW + 8)  = w1;
#endif
  };

  stage(0, 0);
  int buf = 0;

  for (int k0 = 0; k0 < K; k0 += 32) {
#if HAVE_ASYNC_LDS
    wait_async0();
#endif
    __syncthreads();  // current buffer fully staged; prior reads of other buf done
    if (k0 + 32 < K) stage(k0 + 32, buf ^ 1);

    const __bf16* bA = sA[buf];
    const __bf16* bW = sW[buf];

    v16bf af[4];
    #pragma unroll
    for (int rt = 0; rt < 4; rt++) {
      int r = wm * 64 + rt * 16 + l16;
      v8bf lo = *(const v8bf*)(bA + r * 40 + khalf * 8);
      v8bf hi = *(const v8bf*)(bA + r * 40 + 16 + khalf * 8);
      af[rt] = cat8(lo, hi);
    }
    v16bf bfv[4];
    #pragma unroll
    for (int ct = 0; ct < 4; ct++) {
      int c = wn * 64 + ct * 16 + l16;
      v8bf lo = *(const v8bf*)(bW + c * 40 + khalf * 8);
      v8bf hi = *(const v8bf*)(bW + c * 40 + 16 + khalf * 8);
      bfv[ct] = cat8(lo, hi);
    }
    #pragma unroll
    for (int rt = 0; rt < 4; rt++)
      #pragma unroll
      for (int ct = 0; ct < 4; ct++)
        acc[rt][ct] = WMMA_BF16(af[rt], bfv[ct], acc[rt][ct]);

    buf ^= 1;
  }

  #pragma unroll
  for (int rt = 0; rt < 4; rt++) {
    #pragma unroll
    for (int ct = 0; ct < 4; ct++) {
      const int col = colBase + wn * 64 + ct * 16 + l16;
      const float bv = bias[col];
      #pragma unroll
      for (int vi = 0; vi < 8; vi++) {
        const int row = rowBase + wm * 64 + rt * 16 + vi + 8 * khalf;
        const size_t idx = (size_t)row * N + col;
        float v = acc[rt][ct][vi] + bv;
        if constexpr (MODE == 1) v = fmaxf(v, 0.f) + res[idx];
        if constexpr (HASF) outF[idx] = v;
        if constexpr (HASB) outB[idx] = (__bf16)v;
      }
    }
  }
}

// ---------------------------------------------------------------- attention
// Flash-attention per (b,h). Block: 256 thr (8 waves), 128 queries/block,
// each wave owns 16 queries. Streams 32-key blocks through LDS.
// X0 = softmax(Q_h K_h^T / 32) V_h + Qp   (fp32 out [B,NQ,DV])
__global__ void __launch_bounds__(256, 2)
k_attn(const __bf16* __restrict__ Qb, const __bf16* __restrict__ Kb,
       const __bf16* __restrict__ Vb, const float* __restrict__ Qp,
       float* __restrict__ X0) {
  const int D = 1024, HD = 128, NK = 1024;
  __shared__ alignas(16) __bf16 sK[32 * 136];
  __shared__ alignas(16) __bf16 sVt[128 * 40];
  __shared__ alignas(16) float  sS[8][16 * 36];

  const int bh = blockIdx.x;  // 0..63
  const int b  = bh >> 3;
  const int h  = bh & 7;

  const int tid   = threadIdx.x;
  const int lane  = tid & 31;
  const int wave  = tid >> 5;
  const int l16   = lane & 15;
  const int khalf = lane >> 4;
  const int q0    = blockIdx.y * 128 + wave * 16;

  const size_t batchOff = (size_t)b * NK * D;

  // Q fragments (A layout): 16 queries x 128 dims = 4 x (16x32)
  v16bf qf[4];
  {
    const __bf16* qp = Qb + batchOff + (size_t)(q0 + l16) * D + h * HD;
    #pragma unroll
    for (int kc = 0; kc < 4; kc++) {
      v8bf lo = *(const v8bf*)(qp + kc * 32 + khalf * 8);
      v8bf hi = *(const v8bf*)(qp + kc * 32 + 16 + khalf * 8);
      qf[kc] = cat8(lo, hi);
    }
  }

  v8f oacc[8];
  float run_m[8], run_l[8];
  #pragma unroll
  for (int n = 0; n < 8; n++)
    #pragma unroll
    for (int e = 0; e < 8; e++) oacc[n][e] = 0.f;
  #pragma unroll
  for (int e = 0; e < 8; e++) { run_m[e] = -1e30f; run_l[e] = 0.f; }

  const int skey = tid >> 3;  // 0..31
  const int sseg = tid & 7;   // 0..7 (16 elems)

  for (int kb = 0; kb < NK; kb += 32) {
    // ---- stage K rows and V transposed into LDS
    const __bf16* kg = Kb + batchOff + (size_t)(kb + skey) * D + h * HD + sseg * 16;
    const __bf16* vg = Vb + batchOff + (size_t)(kb + skey) * D + h * HD + sseg * 16;
    v8bf k0v = *(const v8bf*)kg;  v8bf k1v = *(const v8bf*)(kg + 8);
    v8bf v0v = *(const v8bf*)vg;  v8bf v1v = *(const v8bf*)(vg + 8);
    __syncthreads();
    *(v8bf*)(sK + skey * 136 + sseg * 16)     = k0v;
    *(v8bf*)(sK + skey * 136 + sseg * 16 + 8) = k1v;
    #pragma unroll
    for (int e = 0; e < 8; e++) {
      sVt[(sseg * 16 + e)     * 40 + skey] = v0v[e];
      sVt[(sseg * 16 + 8 + e) * 40 + skey] = v1v[e];
    }
    __syncthreads();

    // ---- S = Q K^T for two 16-key column tiles
    v8f sacc[2];
    #pragma unroll
    for (int c = 0; c < 2; c++) {
      #pragma unroll
      for (int e = 0; e < 8; e++) sacc[c][e] = 0.f;
      #pragma unroll
      for (int kc = 0; kc < 4; kc++) {
        int key = c * 16 + l16;
        v8bf lo = *(const v8bf*)(sK + key * 136 + kc * 32 + khalf * 8);
        v8bf hi = *(const v8bf*)(sK + key * 136 + kc * 32 + 16 + khalf * 8);
        sacc[c] = WMMA_BF16(qf[kc], cat8(lo, hi), sacc[c]);
      }
    }
    const float scale = 0.03125f;  // 1/sqrt(1024)
    #pragma unroll
    for (int c = 0; c < 2; c++)
      #pragma unroll
      for (int e = 0; e < 8; e++) sacc[c][e] *= scale;

    // ---- online softmax (rows m = e + 8*khalf; reduce over 16-lane halves)
    float nm[8], alpha[8];
    #pragma unroll
    for (int e = 0; e < 8; e++) {
      float m = fmaxf(sacc[0][e], sacc[1][e]);
      for (int off = 1; off < 16; off <<= 1) m = fmaxf(m, __shfl_xor(m, off, 32));
      nm[e] = fmaxf(run_m[e], m);
      alpha[e] = __expf(run_m[e] - nm[e]);
      run_m[e] = nm[e];
    }
    #pragma unroll
    for (int e = 0; e < 8; e++) {
      sacc[0][e] = __expf(sacc[0][e] - nm[e]);
      sacc[1][e] = __expf(sacc[1][e] - nm[e]);
      float s = sacc[0][e] + sacc[1][e];
      for (int off = 1; off < 16; off <<= 1) s += __shfl_xor(s, off, 32);
      run_l[e] = run_l[e] * alpha[e] + s;
    }
    #pragma unroll
    for (int n = 0; n < 8; n++)
      #pragma unroll
      for (int e = 0; e < 8; e++) oacc[n][e] *= alpha[e];

    // ---- re-layout P (C layout -> A layout) through per-wave LDS scratch
    float* ss = sS[wave];
    #pragma unroll
    for (int c = 0; c < 2; c++)
      #pragma unroll
      for (int e = 0; e < 8; e++)
        ss[(e + 8 * khalf) * 36 + c * 16 + l16] = sacc[c][e];

    v16bf pf;
    #pragma unroll
    for (int j = 0; j < 8; j++) pf[j]     = (__bf16)ss[l16 * 36 + khalf * 8 + j];
    #pragma unroll
    for (int j = 0; j < 8; j++) pf[8 + j] = (__bf16)ss[l16 * 36 + 16 + khalf * 8 + j];

    // ---- O += P V
    #pragma unroll
    for (int n = 0; n < 8; n++) {
      int dcol = n * 16 + l16;
      v8bf lo = *(const v8bf*)(sVt + dcol * 40 + khalf * 8);
      v8bf hi = *(const v8bf*)(sVt + dcol * 40 + 16 + khalf * 8);
      oacc[n] = WMMA_BF16(pf, cat8(lo, hi), oacc[n]);
    }
  }

  // ---- epilogue: normalize + Qp residual
  #pragma unroll
  for (int n = 0; n < 8; n++) {
    int dcol = h * HD + n * 16 + l16;
    #pragma unroll
    for (int e = 0; e < 8; e++) {
      int q = q0 + e + 8 * khalf;
      size_t idx = batchOff + (size_t)q * D + dcol;
      X0[idx] = oacc[n][e] / run_l[e] + Qp[idx];
    }
  }
}

// ---------------------------------------------------------------- layernorm
template <bool HASB>
__global__ void __launch_bounds__(256)
k_ln(const float* __restrict__ in, const float* __restrict__ g,
     const float* __restrict__ bt, float* __restrict__ outF,
     __bf16* __restrict__ outB) {
  const int D = 1024;
  __shared__ float red[16];
  size_t row = blockIdx.x;
  const float* x = in + row * D;
  int tid = threadIdx.x;
  float v[4];
  float s = 0.f, s2 = 0.f;
  #pragma unroll
  for (int i = 0; i < 4; i++) {
    v[i] = x[tid + 256 * i];
    s += v[i]; s2 += v[i] * v[i];
  }
  for (int off = 1; off < 32; off <<= 1) {
    s  += __shfl_xor(s, off, 32);
    s2 += __shfl_xor(s2, off, 32);
  }
  int wv = tid >> 5, lane = tid & 31;
  if (lane == 0) { red[wv] = s; red[8 + wv] = s2; }
  __syncthreads();
  if (tid < 32) {
    float a  = (tid < 8) ? red[tid] : 0.f;
    float a2 = (tid < 8) ? red[8 + tid] : 0.f;
    for (int off = 1; off < 8; off <<= 1) {
      a  += __shfl_xor(a, off, 32);
      a2 += __shfl_xor(a2, off, 32);
    }
    if (tid == 0) { red[0] = a; red[1] = a2; }
  }
  __syncthreads();
  float mean = red[0] / D;
  float var  = red[1] / D - mean * mean;
  float rstd = rsqrtf(var + 1e-5f);
  #pragma unroll
  for (int i = 0; i < 4; i++) {
    int c = tid + 256 * i;
    float y = (v[i] - mean) * rstd * g[c] + bt[c];
    outF[row * D + c] = y;
    if constexpr (HASB) outB[row * D + c] = (__bf16)y;
  }
}

// ---------------------------------------------------------------- launch
extern "C" void kernel_launch(void* const* d_in, const int* in_sizes, int n_in,
                              void* d_out, int out_size, void* d_ws, size_t ws_size,
                              hipStream_t stream) {
  const float* Q  = (const float*)d_in[0];
  const float* K  = (const float*)d_in[1];
  const float* Wq = (const float*)d_in[2];
  const float* bq = (const float*)d_in[3];
  const float* Wk = (const float*)d_in[4];
  const float* bk = (const float*)d_in[5];
  const float* Wv = (const float*)d_in[6];
  const float* bv = (const float*)d_in[7];
  const float* Wo = (const float*)d_in[8];
  const float* bo = (const float*)d_in[9];
  const float* g0 = (const float*)d_in[10];
  const float* b0 = (const float*)d_in[11];
  const float* g1 = (const float*)d_in[12];
  const float* b1 = (const float*)d_in[13];
  float* out = (float*)d_out;

  char* ws = (char*)d_ws;
  const size_t MB = 1024ull * 1024ull;
  __bf16* Qbf = (__bf16*)(ws);             // 16 MB
  __bf16* Kbf = (__bf16*)(ws + 16 * MB);   // 16 MB
  __bf16* Wqb = (__bf16*)(ws + 32 * MB);   //  2 MB
  __bf16* Wkb = (__bf16*)(ws + 34 * MB);
  __bf16* Wvb = (__bf16*)(ws + 36 * MB);
  __bf16* Wob = (__bf16*)(ws + 38 * MB);
  float*  Qp  = (float*) (ws + 40 * MB);   // 32 MB
  __bf16* Qpb = (__bf16*)(ws + 72 * MB);   // 16 MB
  __bf16* Kpb = (__bf16*)(ws + 88 * MB);   // 16 MB
  __bf16* Vpb = (__bf16*)(ws + 104 * MB);  // 16 MB
  float*  X0  = (float*) (ws + 120 * MB);  // 32 MB
  __bf16* X1b = (__bf16*)(ws);             // reuse Qbf (dead after projections)
  float*  X2  = (float*) (ws + 40 * MB);   // reuse Qp (dead after attention)

  const int nQK = 8 * 1024 * 1024;
  const int nW  = 1024 * 1024;

  k_f32_to_bf16<<<dim3(2048), dim3(256), 0, stream>>>(Q,  Qbf, nQK);
  k_f32_to_bf16<<<dim3(2048), dim3(256), 0, stream>>>(K,  Kbf, nQK);
  k_f32_to_bf16<<<dim3(512),  dim3(256), 0, stream>>>(Wq, Wqb, nW);
  k_f32_to_bf16<<<dim3(512),  dim3(256), 0, stream>>>(Wk, Wkb, nW);
  k_f32_to_bf16<<<dim3(512),  dim3(256), 0, stream>>>(Wv, Wvb, nW);
  k_f32_to_bf16<<<dim3(512),  dim3(256), 0, stream>>>(Wo, Wob, nW);

  dim3 gg(1024 / 128, 8192 / 256);  // (8, 32)
  k_gemm<0, true, true><<<gg, dim3(256), 0, stream>>>(
      Qbf, Wqb, bq, nullptr, Qp, Qpb, 8192, 1024, 1024);
  k_gemm<0, false, true><<<gg, dim3(256), 0, stream>>>(
      Kbf, Wkb, bk, nullptr, nullptr, Kpb, 8192, 1024, 1024);
  k_gemm<0, false, true><<<gg, dim3(256), 0, stream>>>(
      Kbf, Wvb, bv, nullptr, nullptr, Vpb, 8192, 1024, 1024);

  k_attn<<<dim3(64, 8), dim3(256), 0, stream>>>(Qpb, Kpb, Vpb, Qp, X0);

  k_ln<true><<<dim3(8192), dim3(256), 0, stream>>>(X0, g0, b0, X0, X1b);

  k_gemm<1, true, false><<<gg, dim3(256), 0, stream>>>(
      X1b, Wob, bo, X0, X2, nullptr, 8192, 1024, 1024);

  k_ln<false><<<dim3(8192), dim3(256), 0, stream>>>(X2, g1, b1, out, nullptr);
}
